// MistralMoLoraLayer_55052890800658
// MI455X (gfx1250) — compile-verified, matrix operand order checked by reading, not run on hardware
//
#include <hip/hip_runtime.h>
#include <hip/hip_bf16.h>
#include <stdint.h>

// ---- problem constants (from reference) ----
#define T_TOK    2048      // B*S
#define DMODEL   2048
#define DHIDDEN  5632
#define NEXPERT  8
#define TOPK     2
#define RANK     16
#define ALPHA    2.0f

typedef __bf16 bf16;
typedef __attribute__((ext_vector_type(16))) __bf16 v16bf;
typedef __attribute__((ext_vector_type(8)))  __bf16 bf16x8;
typedef __attribute__((ext_vector_type(8)))  float  v8f;
typedef __attribute__((ext_vector_type(4)))  int    v4i;

// ---- optional gfx1250 async global->LDS path (probe via __has_builtin) ----
#if defined(__has_builtin)
#  if __has_builtin(__builtin_amdgcn_global_load_async_to_lds_b128)
#    define HAVE_ASYNC_LDS 1
#  endif
#endif
#ifndef HAVE_ASYNC_LDS
#  define HAVE_ASYNC_LDS 0
#endif

__device__ __forceinline__ void async_cp16(const bf16* g, bf16* l){
#if HAVE_ASYNC_LDS
  // builtin signature: (int4 AS1*, int4 AS3*, imm offset, imm cpol)
  __builtin_amdgcn_global_load_async_to_lds_b128(
      (__attribute__((address_space(1))) v4i*)(v4i*)(void*)(bf16*)g,
      (__attribute__((address_space(3))) v4i*)(v4i*)(void*)l, 0, 0);
#else
  *(bf16x8*)l = *(const bf16x8*)g;
#endif
}
__device__ __forceinline__ void async_wait(){
#if HAVE_ASYNC_LDS
#  if __has_builtin(__builtin_amdgcn_s_wait_asynccnt)
  __builtin_amdgcn_s_wait_asynccnt(0);
#  else
  asm volatile("s_wait_asynccnt 0x0" ::: "memory");
#  endif
#endif
}

__device__ __forceinline__ bf16 f2bf(float f){
  union { float f; uint32_t u; } v; v.f = f;
  uint32_t r = v.u + 0x7FFFu + ((v.u >> 16) & 1u);   // round-to-nearest-even
  uint16_t h = (uint16_t)(r >> 16);
  return __builtin_bit_cast(bf16, h);
}
__device__ __forceinline__ float bf2f(bf16 b){
  uint16_t h = __builtin_bit_cast(uint16_t, b);
  union { uint32_t u; float f; } v; v.u = ((uint32_t)h) << 16;
  return v.f;
}

// ---------------- fp32 -> bf16 conversion ----------------
__global__ void cvt_f32_bf16(const float* __restrict__ in, bf16* __restrict__ out, int n){
  int i = blockIdx.x * 256 + threadIdx.x;
  if (i < n) out[i] = f2bf(in[i]);
}

// ---------------- WMMA fragment load --------------------
// 16-bit A/B 16x32 layout: lane l -> row (l&15); K-half (l>>4):
// two contiguous 8-elem (16B) chunks at k = khalf*8 and k = 16 + khalf*8.
// Works for global or LDS base pointers (addrspace inferred at call site).
__device__ __forceinline__ v16bf load_frag(const bf16* __restrict__ base, int ld, int lane){
  int r  = lane & 15;
  int kh = (lane >> 4) << 3;
  const bf16* p = base + (size_t)r * ld + kh;
  bf16x8 lo = *(const bf16x8*)(p);
  bf16x8 hi = *(const bf16x8*)(p + 16);
  return __builtin_shufflevector(lo, hi, 0,1,2,3,4,5,6,7,8,9,10,11,12,13,14,15);
}

#define WMMA_BF16(a,b,c) __builtin_amdgcn_wmma_f32_16x16x32_bf16(false, a, false, b, (short)0, c, false, false)

// =========================================================
// Big LDS-staged WMMA GEMM: C[M,N] = A[M,K] * B[N,K]^T
// macro tile 64(M) x 256(N); 8 waves as 2(M) x 4(N); wave tile 32x64.
// K staged in 64-wide slabs through LDS (async copy when available).
// =========================================================
#define KSTEP 64
#define LDP   72              // padded LDS row (elements) -> conflict-free b128 reads

__device__ __forceinline__ void gemm_core(const bf16* __restrict__ Ablk,
                                          const bf16* __restrict__ Bblk,
                                          int K, bf16* smA, bf16* smB,
                                          int wm, int wn, int lane, int tid,
                                          v8f c[2][4]){
  for (int k0 = 0; k0 < K; k0 += KSTEP){
    __syncthreads();                       // previous slab fully consumed
    #pragma unroll
    for (int i = 0; i < 2; ++i){           // A slab: 64 rows x 64 k = 512 chunks
      int cc = tid + i*256;
      int row = cc >> 3, kc = (cc & 7) << 3;
      async_cp16(Ablk + (size_t)row * K + k0 + kc, smA + row*LDP + kc);
    }
    #pragma unroll
    for (int i = 0; i < 8; ++i){           // B slab: 256 rows x 64 k = 2048 chunks
      int cc = tid + i*256;
      int row = cc >> 3, kc = (cc & 7) << 3;
      async_cp16(Bblk + (size_t)row * K + k0 + kc, smB + row*LDP + kc);
    }
    async_wait();
    __syncthreads();                       // slab visible to all waves
    #pragma unroll
    for (int ks = 0; ks < KSTEP; ks += 32){
      v16bf a0 = load_frag(smA + (wm*32 +  0)*LDP + ks, LDP, lane);
      v16bf a1 = load_frag(smA + (wm*32 + 16)*LDP + ks, LDP, lane);
      v16bf b0 = load_frag(smB + (wn*64 +  0)*LDP + ks, LDP, lane);
      v16bf b1 = load_frag(smB + (wn*64 + 16)*LDP + ks, LDP, lane);
      v16bf b2 = load_frag(smB + (wn*64 + 32)*LDP + ks, LDP, lane);
      v16bf b3 = load_frag(smB + (wn*64 + 48)*LDP + ks, LDP, lane);
      c[0][0] = WMMA_BF16(a0, b0, c[0][0]);
      c[0][1] = WMMA_BF16(a0, b1, c[0][1]);
      c[0][2] = WMMA_BF16(a0, b2, c[0][2]);
      c[0][3] = WMMA_BF16(a0, b3, c[0][3]);
      c[1][0] = WMMA_BF16(a1, b0, c[1][0]);
      c[1][1] = WMMA_BF16(a1, b1, c[1][1]);
      c[1][2] = WMMA_BF16(a1, b2, c[1][2]);
      c[1][3] = WMMA_BF16(a1, b3, c[1][3]);
    }
  }
}

template<int OUT_BF16>
__global__ __launch_bounds__(256)
void gemm_big(const bf16* __restrict__ A, const bf16* __restrict__ B,
              float* __restrict__ Cf, bf16* __restrict__ Cb, int N, int K){
  __shared__ bf16 smA[64  * LDP];
  __shared__ bf16 smB[256 * LDP];
  const int tid  = threadIdx.x;
  const int lane = tid & 31;
  const int wave = tid >> 5;
  const int wm = wave & 1, wn = wave >> 1;
  const int mblk = blockIdx.y * 64;
  const int nblk = blockIdx.x * 256;
  v8f c[2][4] = {};
  gemm_core(A + (size_t)mblk * K, B + (size_t)nblk * K, K, smA, smB, wm, wn, lane, tid, c);
  const int n  = lane & 15;
  const int mb = (lane >> 4) << 3;
  #pragma unroll
  for (int i = 0; i < 2; ++i){
    #pragma unroll
    for (int v = 0; v < 8; ++v){
      size_t row = (size_t)(mblk + wm*32 + i*16 + mb + v) * N;
      #pragma unroll
      for (int j = 0; j < 4; ++j){
        size_t idx = row + nblk + wn*64 + j*16 + n;
        if (OUT_BF16) Cb[idx] = f2bf(c[i][j][v]);
        else          Cf[idx] = c[i][j][v];
      }
    }
  }
}

// ---- down GEMM (big tile) + fused LoRA-down epilogue -> d_out ----
__global__ __launch_bounds__(256)
void gemm_down_big(const bf16* __restrict__ A, const bf16* __restrict__ B,
                   const float* __restrict__ Qs, const int* __restrict__ sel,
                   const float* __restrict__ downB, float* __restrict__ out){
  const int K = DHIDDEN, N = DMODEL;
  __shared__ bf16 smA[64  * LDP];
  __shared__ bf16 smB[256 * LDP];
  const int tid  = threadIdx.x;
  const int lane = tid & 31;
  const int wave = tid >> 5;
  const int wm = wave & 1, wn = wave >> 1;
  const int mblk = blockIdx.y * 64;
  const int nblk = blockIdx.x * 256;
  v8f c[2][4] = {};
  gemm_core(A + (size_t)mblk * K, B + (size_t)nblk * K, K, smA, smB, wm, wn, lane, tid, c);
  const int n  = lane & 15;
  const int mb = (lane >> 4) << 3;
  #pragma unroll
  for (int i = 0; i < 2; ++i){
    for (int v = 0; v < 8; ++v){
      const int t = mblk + wm*32 + i*16 + mb + v;
      float l[4] = {0.f, 0.f, 0.f, 0.f};
      #pragma unroll
      for (int k2 = 0; k2 < TOPK; ++k2){
        const int e = sel[t*2 + k2];
        const float* q = Qs + ((size_t)t*2 + k2)*RANK;
        #pragma unroll
        for (int j = 0; j < 4; ++j){
          const int d = nblk + wn*64 + j*16 + n;
          const float* bb = downB + ((size_t)e*N + d)*RANK;
          float s = 0.f;
          #pragma unroll
          for (int r = 0; r < RANK; ++r) s += q[r]*bb[r];
          l[j] += s;
        }
      }
      #pragma unroll
      for (int j = 0; j < 4; ++j){
        const int d = nblk + wn*64 + j*16 + n;
        out[(size_t)t*N + d] = c[i][j][v] + ALPHA*l[j];
      }
    }
  }
}

// ---------------- small-N GEMM (LoRA-A, N=128): direct global loads --------
template<int OUT_BF16>
__global__ void gemm_small(const bf16* __restrict__ A, const bf16* __restrict__ B,
                           float* __restrict__ Cf, bf16* __restrict__ Cb,
                           int N, int K){
  const int lane = threadIdx.x & 31;
  const int wave = threadIdx.x >> 5;
  const int wm = wave & 3;
  const int wn = wave >> 2;
  const int m0 = blockIdx.y * 64 + wm * 16;
  const int n0 = blockIdx.x * 64 + wn * 32;
  const bf16* Ab  = A + (size_t)m0 * K;
  const bf16* Bb0 = B + (size_t)n0 * K;
  const bf16* Bb1 = B + (size_t)(n0 + 16) * K;
  v8f c0 = {}; v8f c1 = {};
  for (int k = 0; k < K; k += 32){
    __builtin_prefetch(Ab + k + 512, 0, 1);           // global_prefetch_b8
    v16bf a  = load_frag(Ab  + k, K, lane);
    v16bf b0 = load_frag(Bb0 + k, K, lane);
    v16bf b1 = load_frag(Bb1 + k, K, lane);
    c0 = WMMA_BF16(a, b0, c0);
    c1 = WMMA_BF16(a, b1, c1);
  }
  const int n  = lane & 15;
  const int mb = (lane >> 4) << 3;
  for (int v = 0; v < 8; ++v){
    size_t row = (size_t)(m0 + mb + v) * N;
    if (OUT_BF16){
      Cb[row + n0 + n]      = f2bf(c0[v]);
      Cb[row + n0 + 16 + n] = f2bf(c1[v]);
    } else {
      Cf[row + n0 + n]      = c0[v];
      Cf[row + n0 + 16 + n] = c1[v];
    }
  }
}

// ---------------- router: logits + top-2 -----------------
__global__ void router_topk(const float* __restrict__ x, const float* __restrict__ gw,
                            float* __restrict__ topv, int* __restrict__ sel){
  const int lane = threadIdx.x & 31;
  const int t = blockIdx.x * 8 + (threadIdx.x >> 5);
  const float* xr = x + (size_t)t * DMODEL;
  float acc[NEXPERT];
  #pragma unroll
  for (int e = 0; e < NEXPERT; ++e) acc[e] = 0.f;
  for (int d = lane; d < DMODEL; d += 32){
    float xv = xr[d];
    #pragma unroll
    for (int e = 0; e < NEXPERT; ++e) acc[e] += xv * gw[e * DMODEL + d];
  }
  #pragma unroll
  for (int e = 0; e < NEXPERT; ++e)
    for (int off = 16; off > 0; off >>= 1) acc[e] += __shfl_xor(acc[e], off, 32);
  if (lane == 0){
    int i0 = 0; float v0 = acc[0];
    #pragma unroll
    for (int e = 1; e < NEXPERT; ++e) if (acc[e] > v0){ v0 = acc[e]; i0 = e; }
    int i1 = (i0 == 0) ? 1 : 0; float v1 = acc[i1];
    #pragma unroll
    for (int e = 0; e < NEXPERT; ++e) if (e != i0 && acc[e] > v1){ v1 = acc[e]; i1 = e; }
    topv[t*2]   = v0; topv[t*2+1] = v1;
    sel[t*2]    = i0; sel[t*2+1]  = i1;
  }
}

// ----- faithful softmax over SEQUENCE axis per (b,k) -----
__global__ void seq_softmax(const float* __restrict__ topv, float* __restrict__ cw){
  __shared__ float red[256];
  const int k = blockIdx.x, b = blockIdx.y, tid = threadIdx.x;
  const int S = 1024;
  float m = -3.4e38f;
  for (int s = tid; s < S; s += 256)
    m = fmaxf(m, topv[((size_t)(b*S + s))*2 + k]);
  red[tid] = m; __syncthreads();
  for (int o = 128; o > 0; o >>= 1){ if (tid < o) red[tid] = fmaxf(red[tid], red[tid+o]); __syncthreads(); }
  m = red[0]; __syncthreads();
  float sum = 0.f;
  for (int s = tid; s < S; s += 256)
    sum += __expf(topv[((size_t)(b*S + s))*2 + k] - m);
  red[tid] = sum; __syncthreads();
  for (int o = 128; o > 0; o >>= 1){ if (tid < o) red[tid] += red[tid+o]; __syncthreads(); }
  const float inv = 1.f / red[0];
  for (int s = tid; s < S; s += 256){
    size_t idx = ((size_t)(b*S + s))*2 + k;
    cw[idx] = __expf(topv[idx] - m) * inv;
  }
}

// ------- per-element expert mix (only top-2 matter) -------
__global__ void expert_mix(const bf16* __restrict__ U, const bf16* __restrict__ G,
                           const float* __restrict__ Pu, const float* __restrict__ Pg,
                           const float* __restrict__ upB, const float* __restrict__ gateB,
                           const float* __restrict__ cw, const int* __restrict__ sel,
                           bf16* __restrict__ hmix, bf16* __restrict__ ch0, bf16* __restrict__ ch1){
  const int t = blockIdx.y;
  const int h = blockIdx.x * 256 + threadIdx.x;
  const size_t idx = (size_t)t * DHIDDEN + h;
  const float u = bf2f(U[idx]);
  const float g = bf2f(G[idx]);
  float acc = 0.f;
  #pragma unroll
  for (int k = 0; k < TOPK; ++k){
    const int e   = sel[t*2 + k];     // uniform within block -> scalarized
    const float w = cw[t*2 + k];
    const float* pu = Pu + (size_t)t*(NEXPERT*RANK) + e*RANK;
    const float* pg = Pg + (size_t)t*(NEXPERT*RANK) + e*RANK;
    const float* bu = upB   + ((size_t)e*DHIDDEN + h)*RANK;
    const float* bg = gateB + ((size_t)e*DHIDDEN + h)*RANK;
    float du = 0.f, dg = 0.f;
    #pragma unroll
    for (int r = 0; r < RANK; ++r){ du += pu[r]*bu[r]; dg += pg[r]*bg[r]; }
    const float u2 = u + ALPHA*du;
    const float g2 = g + ALPHA*dg;
    const float he = (u2 / (1.f + __expf(-u2))) * g2;   // silu(u2) * g2
    const float chv = w * he;
    acc += chv;
    if (k == 0) ch0[idx] = f2bf(chv); else ch1[idx] = f2bf(chv);
  }
  hmix[idx] = f2bf(acc);
}

// -------- Q[t,k,:] = ch_k[t,:] @ down_A[e]^T (rank 16) ----
__global__ void qproj(const bf16* __restrict__ ch0, const bf16* __restrict__ ch1,
                      const int* __restrict__ sel, const float* __restrict__ downA,
                      float* __restrict__ Qs){
  const int k = blockIdx.x;
  const int t = blockIdx.y;
  const int lane = threadIdx.x;
  const int e = sel[t*2 + k];
  const bf16* ch = (k ? ch1 : ch0) + (size_t)t * DHIDDEN;
  const float* dA = downA + (size_t)e * RANK * DHIDDEN;
  float acc[RANK];
  #pragma unroll
  for (int r = 0; r < RANK; ++r) acc[r] = 0.f;
  for (int h = lane; h < DHIDDEN; h += 32){
    const float c = bf2f(ch[h]);
    #pragma unroll
    for (int r = 0; r < RANK; ++r) acc[r] += c * dA[(size_t)r*DHIDDEN + h];
  }
  #pragma unroll
  for (int r = 0; r < RANK; ++r)
    for (int off = 16; off > 0; off >>= 1) acc[r] += __shfl_xor(acc[r], off, 32);
  if (lane == 0){
    float* q = Qs + ((size_t)t*2 + k)*RANK;
    #pragma unroll
    for (int r = 0; r < RANK; ++r) q[r] = acc[r];
  }
}

extern "C" void kernel_launch(void* const* d_in, const int* in_sizes, int n_in,
                              void* d_out, int out_size, void* d_ws, size_t ws_size,
                              hipStream_t stream){
  const float* x      = (const float*)d_in[0];
  const float* gate_w = (const float*)d_in[1];
  const float* w_up   = (const float*)d_in[2];
  const float* w_gate = (const float*)d_in[3];
  const float* w_down = (const float*)d_in[4];
  const float* up_A   = (const float*)d_in[5];
  const float* up_B   = (const float*)d_in[6];
  const float* gate_A = (const float*)d_in[7];
  const float* gate_B = (const float*)d_in[8];
  const float* down_A = (const float*)d_in[9];
  const float* down_B = (const float*)d_in[10];
  float* out = (float*)d_out;

  char* ws = (char*)d_ws;
  size_t off = 0;
  auto alloc = [&](size_t bytes)->char*{
    char* p = ws + off;
    off = (off + bytes + 255) & ~(size_t)255;
    return p;
  };
  const size_t TD = (size_t)T_TOK * DMODEL;
  const size_t HD = (size_t)DHIDDEN * DMODEL;
  const size_t TH = (size_t)T_TOK * DHIDDEN;

  bf16*  xb     = (bf16*)alloc(TD * 2);
  bf16*  wub    = (bf16*)alloc(HD * 2);
  bf16*  wgb    = (bf16*)alloc(HD * 2);
  bf16*  wdb    = (bf16*)alloc(HD * 2);
  bf16*  upAb   = (bf16*)alloc((size_t)NEXPERT*RANK*DMODEL * 2);
  bf16*  gateAb = (bf16*)alloc((size_t)NEXPERT*RANK*DMODEL * 2);
  bf16*  Ub     = (bf16*)alloc(TH * 2);
  bf16*  Gb     = (bf16*)alloc(TH * 2);
  bf16*  ch1b   = (bf16*)alloc(TH * 2);
  float* Pu     = (float*)alloc((size_t)T_TOK * NEXPERT*RANK * 4);
  float* Pg     = (float*)alloc((size_t)T_TOK * NEXPERT*RANK * 4);
  float* Qs     = (float*)alloc((size_t)T_TOK * TOPK * RANK * 4);
  float* topv   = (float*)alloc((size_t)T_TOK * 2 * 4);
  int*   sel    = (int*)  alloc((size_t)T_TOK * 2 * 4);
  float* cw     = (float*)alloc((size_t)T_TOK * 2 * 4);
  // phase-safe aliases: w_up/w_gate bf16 copies are dead after the U/G GEMMs
  bf16*  hmixb  = wub;
  bf16*  ch0b   = wgb;

  // 1) bf16 conversions
  cvt_f32_bf16<<<(int)((TD + 255)/256), 256, 0, stream>>>(x, xb, (int)TD);
  cvt_f32_bf16<<<(int)((HD + 255)/256), 256, 0, stream>>>(w_up,   wub, (int)HD);
  cvt_f32_bf16<<<(int)((HD + 255)/256), 256, 0, stream>>>(w_gate, wgb, (int)HD);
  cvt_f32_bf16<<<(int)((HD + 255)/256), 256, 0, stream>>>(w_down, wdb, (int)HD);
  const int nA = NEXPERT*RANK*DMODEL;
  cvt_f32_bf16<<<(nA + 255)/256, 256, 0, stream>>>(up_A,   upAb,   nA);
  cvt_f32_bf16<<<(nA + 255)/256, 256, 0, stream>>>(gate_A, gateAb, nA);

  // 2) router + sequence-axis softmax
  router_topk<<<T_TOK/8, 256, 0, stream>>>(x, gate_w, topv, sel);
  seq_softmax<<<dim3(2, 2), 256, 0, stream>>>(topv, cw);

  // 3) LoRA-A projections: P = x @ A^T  (N = E*R = 128)
  gemm_small<0><<<dim3((NEXPERT*RANK)/64, T_TOK/64), 256, 0, stream>>>(xb, upAb,   Pu, nullptr, NEXPERT*RANK, DMODEL);
  gemm_small<0><<<dim3((NEXPERT*RANK)/64, T_TOK/64), 256, 0, stream>>>(xb, gateAb, Pg, nullptr, NEXPERT*RANK, DMODEL);

  // 4) base projections: U = x @ w_up^T, G = x @ w_gate^T (LDS-staged big tiles)
  gemm_big<1><<<dim3(DHIDDEN/256, T_TOK/64), 256, 0, stream>>>(xb, wub, nullptr, Ub, DHIDDEN, DMODEL);
  gemm_big<1><<<dim3(DHIDDEN/256, T_TOK/64), 256, 0, stream>>>(xb, wgb, nullptr, Gb, DHIDDEN, DMODEL);

  // 5) expert mix (top-2 only) -> h_mix, ch0, ch1
  expert_mix<<<dim3(DHIDDEN/256, T_TOK), 256, 0, stream>>>(Ub, Gb, Pu, Pg, up_B, gate_B, cw, sel,
                                                           hmixb, ch0b, ch1b);
  // 6) down-LoRA A projection per (token, slot)
  qproj<<<dim3(TOPK, T_TOK), 32, 0, stream>>>(ch0b, ch1b, sel, down_A, Qs);

  // 7) down GEMM + fused LoRA-down epilogue
  gemm_down_big<<<dim3(DMODEL/256, T_TOK/64), 256, 0, stream>>>(hmixb, wdb, Qs, sel, down_B, out);
}
